// FlashAttention_69269232550200
// MI455X (gfx1250) — matrix operand order repaired and checked by reading, not compiled, not run on hardware
//
#include <hip/hip_runtime.h>

// MI455X / gfx1250: wave32, WMMA 16x16x32 bf16 (fp32 accum), async LDS staging

typedef __attribute__((ext_vector_type(16))) __bf16 v16bf;
typedef __attribute__((ext_vector_type(8)))  float  v8f;

union FragU { v16bf v; uint4 u[2]; };

#define WMMA_BF16(A, Bm, C) \
  __builtin_amdgcn_wmma_f32_16x16x32_bf16(false, (A), false, (Bm), (short)0, (C), false, false)

// ---------------- CDNA5 async global->LDS (CDNA5 ISA) ---
__device__ __forceinline__ void async_load_b128(unsigned lds_off, const void* gaddr) {
  asm volatile("global_load_async_to_lds_b128 %0, %1, off"
               :: "v"(lds_off), "v"((unsigned long long)(uintptr_t)gaddr)
               : "memory");
}
__device__ __forceinline__ void wait_async0() {
  asm volatile("s_wait_asynccnt 0" ::: "memory");
}

// ---------------- fragment loaders (layouts per CDNA5 ISA) ------
// A-frag 16x32: lane m = lane&15; elem i -> K = 16*(i>>3) + 8*half + (i&7)
__device__ __forceinline__ v16bf load_a_frag(const __bf16* base, int ld,
                                             int row0, int k0, int lane) {
  int hlf = lane >> 4, m = lane & 15;
  const __bf16* p = base + (size_t)(row0 + m) * ld + k0 + hlf * 8;
  FragU f;
  f.u[0] = *(const uint4*)p;         // K = 8h .. 8h+7
  f.u[1] = *(const uint4*)(p + 16);  // K = 16+8h .. 16+8h+7
  return f.v;
}

// B-frag 32x16: lane n = lane&15 (output col); elem i -> K = 16*(i>>3) + 8*half + (i&7)
__device__ __forceinline__ v16bf load_b_frag(const __bf16* base, int ld,
                                             int col0, int k0, int lane) {
  int hlf = lane >> 4, n = lane & 15;
  const __bf16* p = base + (size_t)(col0 + n) * ld + k0 + hlf * 8;
  FragU f;
  f.u[0] = *(const uint4*)p;
  f.u[1] = *(const uint4*)(p + 16);
  return f.v;
}

// ---------------- conversion kernels ---------------------------------------
__global__ __launch_bounds__(256) void k_cvt_bf16(const float* __restrict__ in,
                                                  __bf16* __restrict__ out, int n) {
  int i = blockIdx.x * 256 + threadIdx.x;
  if (i < n) out[i] = (__bf16)in[i];
}

// out[n][k] = (bf16) in[k][n]   (2048x2048, LDS-tiled)
__global__ __launch_bounds__(256) void k_transpose_cvt(const float* __restrict__ in,
                                                       __bf16* __restrict__ out) {
  __shared__ float tile[32][33];
  int bx = blockIdx.x * 32, by = blockIdx.y * 32;
  int tx = threadIdx.x & 31, ty = threadIdx.x >> 5;
#pragma unroll
  for (int i = 0; i < 32; i += 8)
    tile[ty + i][tx] = in[(size_t)(by + ty + i) * 2048 + bx + tx];
  __syncthreads();
#pragma unroll
  for (int i = 0; i < 32; i += 8)
    out[(size_t)(bx + ty + i) * 2048 + by + tx] = (__bf16)tile[tx][ty + i];
}

// ---------------- GEMM: out = A[4096x2048] * W  (W given transposed) -------
// Wave tile 32x64 (8 WMMAs / K-step), ping-pong double-buffered K loop.
// mode 0: fp32 row-major; mode 1: bf16 [b][h][s][d]; mode 2: bf16 [b][h][d][s]
__global__ __launch_bounds__(256) void k_gemm(const __bf16* __restrict__ A,
                                              const __bf16* __restrict__ Bt,
                                              const float* __restrict__ bias,
                                              void* __restrict__ outp, int mode) {
  int w = threadIdx.x >> 5, lane = threadIdx.x & 31;
  int wm = w >> 2, wn = w & 3;
  int m0 = blockIdx.y * 64 + wm * 32;
  int n0 = blockIdx.x * 256 + wn * 64;
  v8f acc[2][4] = {};
  v16bf a0[2], b0[4], a1[2], b1[4];

#pragma unroll
  for (int i = 0; i < 2; i++) a0[i] = load_a_frag(A, 2048, m0 + i * 16, 0, lane);
#pragma unroll
  for (int t = 0; t < 4; t++) b0[t] = load_b_frag(Bt, 2048, n0 + t * 16, 0, lane);

  for (int k0 = 0; k0 < 2048; k0 += 64) {
    // stream future A rows into L2
    __builtin_prefetch(A + (size_t)(m0 + (lane & 15)) * 2048 + k0 + 128, 0, 1);
    // stage k0+32 while computing k0
#pragma unroll
    for (int i = 0; i < 2; i++) a1[i] = load_a_frag(A, 2048, m0 + i * 16, k0 + 32, lane);
#pragma unroll
    for (int t = 0; t < 4; t++) b1[t] = load_b_frag(Bt, 2048, n0 + t * 16, k0 + 32, lane);
#pragma unroll
    for (int i = 0; i < 2; i++)
#pragma unroll
      for (int t = 0; t < 4; t++) acc[i][t] = WMMA_BF16(a0[i], b0[t], acc[i][t]);
    if (k0 + 64 < 2048) {  // stage k0+64 while computing k0+32
#pragma unroll
      for (int i = 0; i < 2; i++) a0[i] = load_a_frag(A, 2048, m0 + i * 16, k0 + 64, lane);
#pragma unroll
      for (int t = 0; t < 4; t++) b0[t] = load_b_frag(Bt, 2048, n0 + t * 16, k0 + 64, lane);
    }
#pragma unroll
    for (int i = 0; i < 2; i++)
#pragma unroll
      for (int t = 0; t < 4; t++) acc[i][t] = WMMA_BF16(a1[i], b1[t], acc[i][t]);
  }

  int hlf = lane >> 4, nl = lane & 15;
#pragma unroll
  for (int i = 0; i < 2; i++) {
#pragma unroll
    for (int t = 0; t < 4; t++) {
      int n = n0 + t * 16 + nl;
      float bv = bias[n];
#pragma unroll
      for (int r = 0; r < 8; r++) {
        int m = m0 + i * 16 + r + 8 * hlf;  // D layout: VGPR r = row r+8*half
        float v = acc[i][t][r] + bv;
        int bb = m >> 11, s = m & 2047;     // S = 2048
        int h = n >> 7, d = n & 127;        // HD = 128
        if (mode == 0) {
          ((float*)outp)[(size_t)m * 2048 + n] = v;
        } else if (mode == 1) {
          ((__bf16*)outp)[(((size_t)(bb * 16 + h)) * 2048 + s) * 128 + d] = (__bf16)v;
        } else {
          ((__bf16*)outp)[(((size_t)(bb * 16 + h)) * 128 + d) * 2048 + s] = (__bf16)v;
        }
      }
    }
  }
}

// ---------------- flash attention (global w/ alibi, or local window) -------
// grid.x = B*H*(S/128); 8 waves x 16 q-rows, full HD=128 accumulator.
// K/V^T tiles staged cooperatively via async global->LDS (ASYNCcnt path);
// all per-wave B-fragments then come from LDS (ds_read_b128).
__global__ __launch_bounds__(256) void k_attn(const __bf16* __restrict__ Q,
                                              const __bf16* __restrict__ K,
                                              const __bf16* __restrict__ Vt,
                                              __bf16* __restrict__ ctx,
                                              int is_global) {
  __shared__ __bf16 Kt[32][128];    // [key][d]      8 KB (contiguous in global)
  __shared__ __bf16 Vs[128][32];    // [d][key]      8 KB
  __shared__ __bf16 plds[8][16][32];// per-wave P    8 KB
  int tid = threadIdx.x;
  int w = tid >> 5, lane = tid & 31;
  int hlf = lane >> 4, nl = lane & 15;
  int blk = blockIdx.x;
  int qb = blk & 15;          // 16 q-blocks of 128
  int h  = (blk >> 4) & 15;
  int b  = blk >> 8;
  int q0w = qb * 128 + w * 16;
  int ks   = is_global ? 0 : (q0w & ~255);   // local: 256-token window
  int klen = is_global ? 2048 : 256;

  const __bf16* Qbh = Q  + ((size_t)(b * 16 + h)) * 2048 * 128;
  const __bf16* Kbh = K  + ((size_t)(b * 16 + h)) * 2048 * 128;
  const __bf16* Vbh = Vt + ((size_t)(b * 16 + h)) * 128 * 2048;
  unsigned kt_lds = (unsigned)(uintptr_t)&Kt[0][0];
  unsigned vs_lds = (unsigned)(uintptr_t)&Vs[0][0];

  v16bf qa[4];
#pragma unroll
  for (int dd = 0; dd < 4; dd++) qa[dd] = load_a_frag(Qbh, 128, q0w, dd * 32, lane);

  v8f acc[8] = {};
  float mX[8], ls[8];
#pragma unroll
  for (int r = 0; r < 8; r++) { mX[r] = -1e30f; ls[r] = 0.f; }
  const float scale = 0.08838834764831845f;  // 1/sqrt(128)

  for (int kc = ks; kc < ks + klen; kc += 32) {
    __syncthreads();  // previous iteration's tile reads complete
    {  // stage K tile: 8 KB contiguous; V^T tile: 128 rows x 64 B
      const char* kg = (const char*)(Kbh + (size_t)kc * 128);
#pragma unroll
      for (int is = 0; is < 2; is++) {
        int f = tid * 16 + is * 4096;
        async_load_b128(kt_lds + f, kg + f);
      }
#pragma unroll
      for (int is = 0; is < 2; is++) {
        int f = tid * 16 + is * 4096;
        int d = f >> 6, cb = f & 63;
        async_load_b128(vs_lds + f, (const char*)(Vbh + (size_t)d * 2048 + kc) + cb);
      }
      __builtin_prefetch((const char*)(Kbh + (size_t)(kc + 32) * 128), 0, 1);
    }
    wait_async0();
    __syncthreads();  // tiles visible to all waves

    v8f s0 = {}, s1 = {};
#pragma unroll
    for (int dd = 0; dd < 4; dd++) {  // QK^T: B-frags from LDS K tile
      v16bf kb0 = load_b_frag((const __bf16*)Kt, 128, 0,  dd * 32, lane);
      v16bf kb1 = load_b_frag((const __bf16*)Kt, 128, 16, dd * 32, lane);
      s0 = WMMA_BF16(qa[dd], kb0, s0);
      s1 = WMMA_BF16(qa[dd], kb1, s1);
    }
    float p0[8], p1[8], red[8];
#pragma unroll
    for (int r = 0; r < 8; r++) {
      float a0 = s0[r] * scale, a1 = s1[r] * scale;
      if (is_global) {  // alibi slope 1.0 (all heads)
        float qi = (float)(q0w + r + 8 * hlf);
        a0 -= fabsf(qi - (float)(kc + nl));
        a1 -= fabsf(qi - (float)(kc + 16 + nl));
      }
      p0[r] = a0; p1[r] = a1;
      red[r] = fmaxf(a0, a1);
    }
#pragma unroll
    for (int off = 1; off < 16; off <<= 1)  // row max within 16-lane half
#pragma unroll
      for (int r = 0; r < 8; r++)
        red[r] = fmaxf(red[r], __shfl_xor(red[r], off, 32));
    float cr[8];
#pragma unroll
    for (int r = 0; r < 8; r++) {
      float mn = fmaxf(mX[r], red[r]);
      cr[r] = __expf(mX[r] - mn);
      mX[r] = mn;
      p0[r] = __expf(p0[r] - mn);
      p1[r] = __expf(p1[r] - mn);
      red[r] = p0[r] + p1[r];
    }
#pragma unroll
    for (int off = 1; off < 16; off <<= 1)  // row sum within half
#pragma unroll
      for (int r = 0; r < 8; r++)
        red[r] += __shfl_xor(red[r], off, 32);
#pragma unroll
    for (int r = 0; r < 8; r++) {
      ls[r] = ls[r] * cr[r] + red[r];
#pragma unroll
      for (int dt = 0; dt < 8; dt++) acc[dt][r] *= cr[r];
    }
    // stage P (D-layout) into wave-local LDS slab, re-read as A-fragment
#pragma unroll
    for (int r = 0; r < 8; r++) {
      plds[w][r + 8 * hlf][nl]      = (__bf16)p0[r];
      plds[w][r + 8 * hlf][16 + nl] = (__bf16)p1[r];
    }
    FragU pf;
    const __bf16* pb = &plds[w][nl][hlf * 8];
    pf.u[0] = *(const uint4*)pb;        // keys 8h..8h+7
    pf.u[1] = *(const uint4*)(pb + 16); // keys 16+8h..
#pragma unroll
    for (int dt = 0; dt < 8; dt++) {    // PV: B-frags from LDS V^T tile
      v16bf vb = load_b_frag((const __bf16*)Vs, 32, dt * 16, 0, lane);
      acc[dt] = WMMA_BF16(pf.v, vb, acc[dt]);
    }
  }
  // normalize + store ctx as bf16 [b][s][e], e = h*128 + d
#pragma unroll
  for (int dt = 0; dt < 8; dt++) {
#pragma unroll
    for (int r = 0; r < 8; r++) {
      float o = acc[dt][r] / ls[r];
      int m = q0w + r + 8 * hlf;
      ctx[((size_t)(b * 2048 + m)) * 2048 + h * 128 + dt * 16 + nl] = (__bf16)o;
    }
  }
}

// ---------------- per-token global/local select ----------------------------
__global__ __launch_bounds__(256) void k_mix(const __bf16* __restrict__ cg,
                                             const __bf16* __restrict__ cl,
                                             const int* __restrict__ mask,
                                             __bf16* __restrict__ ctx, int total) {
  int i = blockIdx.x * 256 + threadIdx.x;
  if (i < total) {
    int row = i >> 11;  // E = 2048
    ctx[i] = (mask[row] == 1) ? cg[i] : cl[i];
  }
}

// ---------------- host orchestration ---------------------------------------
extern "C" void kernel_launch(void* const* d_in, const int* in_sizes, int n_in,
                              void* d_out, int out_size, void* d_ws, size_t ws_size,
                              hipStream_t stream) {
  (void)in_sizes; (void)n_in; (void)out_size; (void)ws_size;
  const float* x  = (const float*)d_in[0];
  const int*   gm = (const int*)d_in[1];
  const float* W[7]    = {(const float*)d_in[2],  (const float*)d_in[4],
                          (const float*)d_in[6],  (const float*)d_in[8],
                          (const float*)d_in[10], (const float*)d_in[12],
                          (const float*)d_in[14]};
  const float* bias[7] = {(const float*)d_in[3],  (const float*)d_in[5],
                          (const float*)d_in[7],  (const float*)d_in[9],
                          (const float*)d_in[11], (const float*)d_in[13],
                          (const float*)d_in[15]};

  char* ws = (char*)d_ws;
  size_t off = 0;
  auto alloc = [&](size_t bytes) -> void* {
    void* p = ws + off;
    off += (bytes + 255) & ~(size_t)255;
    return p;
  };
  const size_t ACT = (size_t)4096 * 2048 * 2;  // bf16 activation plane
  __bf16* xb = (__bf16*)alloc(ACT);
  __bf16* wt[7];
  for (int i = 0; i < 7; i++) wt[i] = (__bf16*)alloc((size_t)2048 * 2048 * 2);
  __bf16* Qb   = (__bf16*)alloc(ACT);
  __bf16* Kb   = (__bf16*)alloc(ACT);
  __bf16* Vtb  = (__bf16*)alloc(ACT);
  __bf16* LQb  = (__bf16*)alloc(ACT);
  __bf16* LKb  = (__bf16*)alloc(ACT);
  __bf16* LVtb = (__bf16*)alloc(ACT);
  __bf16* cg   = (__bf16*)alloc(ACT);
  __bf16* cl   = (__bf16*)alloc(ACT);
  __bf16* cmix = (__bf16*)alloc(ACT);

  const int NTOT = 4096 * 2048;
  k_cvt_bf16<<<(NTOT + 255) / 256, 256, 0, stream>>>(x, xb, NTOT);
  for (int i = 0; i < 7; i++)
    k_transpose_cvt<<<dim3(64, 64), 256, 0, stream>>>(W[i], wt[i]);

  dim3 gg(8, 64);  // N/256, M/64
  k_gemm<<<gg, 256, 0, stream>>>(xb, wt[0], bias[0], Qb,   1);
  k_gemm<<<gg, 256, 0, stream>>>(xb, wt[1], bias[1], Kb,   1);
  k_gemm<<<gg, 256, 0, stream>>>(xb, wt[2], bias[2], Vtb,  2);
  k_gemm<<<gg, 256, 0, stream>>>(xb, wt[3], bias[3], LQb,  1);
  k_gemm<<<gg, 256, 0, stream>>>(xb, wt[4], bias[4], LKb,  1);
  k_gemm<<<gg, 256, 0, stream>>>(xb, wt[5], bias[5], LVtb, 2);

  k_attn<<<512, 256, 0, stream>>>(Qb,  Kb,  Vtb,  cg, 1);  // global + alibi
  k_attn<<<512, 256, 0, stream>>>(LQb, LKb, LVtb, cl, 0);  // local windows

  k_mix<<<(NTOT + 255) / 256, 256, 0, stream>>>(cg, cl, gm, cmix, NTOT);

  k_gemm<<<gg, 256, 0, stream>>>(cmix, wt[6], bias[6], d_out, 0);  // fp32 out
}
